// SAGEGraphEmbedding_7773890806281
// MI455X (gfx1250) — matrix operand order repaired and checked by reading, not compiled, not run on hardware
//
#include <hip/hip_runtime.h>
#include <hip/hip_bf16.h>

typedef __attribute__((ext_vector_type(16))) _Float16 v16h;
typedef __attribute__((ext_vector_type(8)))  float    v8f;

#define N_NODES_C 50000
#define N_EDGES_C 500000
#define D_C 128
#define N_GRAPHS_C 512

// ---------------------------------------------------------------------------
// zero fill
// ---------------------------------------------------------------------------
__global__ void zero_f32_kernel(float* __restrict__ p, long n) {
    long i = (long)blockIdx.x * blockDim.x + threadIdx.x;
    if (i < n) p[i] = 0.0f;
}

// ---------------------------------------------------------------------------
// edge scatter: agg[dst] += x[src], cnt[dst] += 1   (one wave32 per edge,
// each lane owns 4 consecutive features -> float4 gather + 4 f32 atomics)
// ---------------------------------------------------------------------------
__global__ __launch_bounds__(256) void scatter_accum_kernel(
    const float* __restrict__ x, const int* __restrict__ src,
    const int* __restrict__ dst, float* __restrict__ agg,
    float* __restrict__ cnt, int n_edges)
{
    long gid  = (long)blockIdx.x * blockDim.x + threadIdx.x;
    int  e    = (int)(gid >> 5);
    int  lane = (int)(gid & 31);
    if (e >= n_edges) return;
    int s = src[e];
    int d = dst[e];
    const float4 v = *reinterpret_cast<const float4*>(x + (size_t)s * D_C + lane * 4);
    float* base = agg + (size_t)d * D_C + lane * 4;
    atomicAdd(base + 0, v.x);
    atomicAdd(base + 1, v.y);
    atomicAdd(base + 2, v.z);
    atomicAdd(base + 3, v.w);
    if (lane == 0) atomicAdd(cnt + d, 1.0f);
}

// ---------------------------------------------------------------------------
// agg /= max(cnt, 1)
// ---------------------------------------------------------------------------
__global__ void mean_div_kernel(float* __restrict__ agg,
                                const float* __restrict__ cnt, int n_nodes) {
    long i = (long)blockIdx.x * blockDim.x + threadIdx.x;
    if (i >= (long)n_nodes * D_C) return;
    float c = cnt[i >> 7];              // D_C == 128
    agg[i] = agg[i] / fmaxf(c, 1.0f);
}

// ---------------------------------------------------------------------------
// WMMA fragment loaders (f32 memory -> f16 fragments, ISA 7.12.2 layouts)
//
// A 16x32 f16: lanes 0-15 row M=lane, K-halves {kb..kb+7, kb+16..kb+23} with
// kb = 0; lanes 16-31 same rows, kb = 8.
// B 32x16 f16: lane owns column N=lane%16, K-halves {kb..kb+7, kb+16..kb+23}, kb = hs*8.
// ---------------------------------------------------------------------------
__device__ inline v16h load_a_frag(const float* __restrict__ A, int row,
                                   int k0, int kb) {
    v16h a;
    const float* p0 = A + (size_t)row * D_C + k0 + kb;
    const float* p1 = p0 + 16;
#pragma unroll
    for (int j = 0; j < 8; ++j) a[j] = (_Float16)p0[j];
#pragma unroll
    for (int j = 0; j < 8; ++j) a[8 + j] = (_Float16)p1[j];
    return a;
}

__device__ inline v16h load_b_frag(const float* __restrict__ W, int col,
                                   int k0, int kb16) {
    // computing y = in @ W^T  =>  B[k, n] = W[n, k]
    v16h b;
    const float* p = W + (size_t)col * D_C + k0 + kb16 / 2;
#pragma unroll
    for (int j = 0; j < 8; ++j) b[j] = (_Float16)p[j];
#pragma unroll
    for (int j = 0; j < 8; ++j) b[8 + j] = (_Float16)p[16 + j];
    return b;
}

// ---------------------------------------------------------------------------
// fused SAGE linear: y = relu(agg @ Wl^T + xin @ Wr^T)
// one wave per 16x16 tile; 8 waves/block cover all 8 N-tiles of one M-tile
// ---------------------------------------------------------------------------
__global__ __launch_bounds__(256) void sage_gemm_relu_kernel(
    const float* __restrict__ agg, const float* __restrict__ xin,
    const float* __restrict__ Wl,  const float* __restrict__ Wr,
    float* __restrict__ y, int n_nodes)
{
    const int wave = threadIdx.x >> 5;     // 0..7  -> N tile
    const int lane = threadIdx.x & 31;
    const int r    = lane & 15;
    const int hs   = lane >> 4;            // lane half-select
    const int m0   = blockIdx.x * 16;
    const int n0   = wave * 16;
    if (m0 >= n_nodes) return;             // block-uniform -> EXEC stays all-1s

    const int kb   = hs * 8;
    const int kb16 = hs * 16;

    v8f c = {};
#pragma unroll
    for (int k0 = 0; k0 < D_C; k0 += 32) {
        v16h aA = load_a_frag(agg, m0 + r, k0, kb);
        v16h bL = load_b_frag(Wl,  n0 + r, k0, kb16);
        c = __builtin_amdgcn_wmma_f32_16x16x32_f16(false, aA, false, bL,
                                                   (short)0, c, false, false);
        v16h aX = load_a_frag(xin, m0 + r, k0, kb);
        v16h bR = load_b_frag(Wr,  n0 + r, k0, kb16);
        c = __builtin_amdgcn_wmma_f32_16x16x32_f16(false, aX, false, bR,
                                                   (short)0, c, false, false);
    }

    // C/D layout: VGPR v -> row m0 + v + hs*8, col n0 + r
#pragma unroll
    for (int v = 0; v < 8; ++v) {
        float val = c[v];
        val = val > 0.0f ? val : 0.0f;     // relu
        y[(size_t)(m0 + v + hs * 8) * D_C + n0 + r] = val;
    }
}

// ---------------------------------------------------------------------------
// segment-max pool of [x1 || x2] by batch id. Values are post-ReLU (>= 0),
// so uint atomicMax on the IEEE bits is order-correct, and out preinit 0
// matches the reference's isfinite guard for empty graphs.
// ---------------------------------------------------------------------------
__global__ void pool_max_kernel(const float* __restrict__ x1,
                                const float* __restrict__ x2,
                                const int* __restrict__ batch,
                                float* __restrict__ out, int n_nodes)
{
    long tid = (long)blockIdx.x * blockDim.x + threadIdx.x;
    if (tid >= (long)n_nodes * 256) return;
    int n = (int)(tid >> 8);
    int d = (int)(tid & 255);
    int g = batch[n];
    float val = (d < D_C) ? x1[(size_t)n * D_C + d]
                          : x2[(size_t)n * D_C + (d - D_C)];
    atomicMax(reinterpret_cast<unsigned int*>(out) + (size_t)g * 256 + d,
              __float_as_uint(val));
}

// ---------------------------------------------------------------------------
// launch
// ---------------------------------------------------------------------------
extern "C" void kernel_launch(void* const* d_in, const int* in_sizes, int n_in,
                              void* d_out, int out_size, void* d_ws, size_t ws_size,
                              hipStream_t stream) {
    const float* x     = (const float*)d_in[0];
    const int*   edge  = (const int*)d_in[1];
    const int*   batch = (const int*)d_in[2];
    const float* Wl1   = (const float*)d_in[3];
    const float* Wr1   = (const float*)d_in[4];
    const float* Wl2   = (const float*)d_in[5];
    const float* Wr2   = (const float*)d_in[6];
    float*       out   = (float*)d_out;

    const int NN = N_NODES_C;
    const int NE = N_EDGES_C;

    // workspace layout (f32): agg[NN*128] cnt[NN] x1[NN*128] x2[NN*128]
    float* agg = (float*)d_ws;
    float* cnt = agg + (size_t)NN * D_C;
    float* x1  = cnt + NN;
    float* x2  = x1 + (size_t)NN * D_C;

    const int* src = edge;
    const int* dst = edge + NE;

    const long aggcnt_n  = (long)NN * D_C + NN;     // agg+cnt contiguous
    const int  zb        = (int)((aggcnt_n + 255) / 256);
    const int  scatter_b = (int)(((long)NE * 32 + 255) / 256);
    const int  div_b     = (int)(((long)NN * D_C + 255) / 256);
    const int  gemm_b    = NN / 16;                 // 3125, exact
    const int  pool_b    = (int)(((long)NN * 256 + 255) / 256);
    const int  outz_b    = (N_GRAPHS_C * 256 + 255) / 256;

    // ---- layer 1 ----
    zero_f32_kernel<<<zb, 256, 0, stream>>>(agg, aggcnt_n);
    scatter_accum_kernel<<<scatter_b, 256, 0, stream>>>(x, src, dst, agg, cnt, NE);
    mean_div_kernel<<<div_b, 256, 0, stream>>>(agg, cnt, NN);
    sage_gemm_relu_kernel<<<gemm_b, 256, 0, stream>>>(agg, x, Wl1, Wr1, x1, NN);

    // ---- layer 2 ----
    zero_f32_kernel<<<zb, 256, 0, stream>>>(agg, aggcnt_n);
    scatter_accum_kernel<<<scatter_b, 256, 0, stream>>>(x1, src, dst, agg, cnt, NE);
    mean_div_kernel<<<div_b, 256, 0, stream>>>(agg, cnt, NN);
    sage_gemm_relu_kernel<<<gemm_b, 256, 0, stream>>>(agg, x1, Wl2, Wr2, x2, NN);

    // ---- JK-cat + segment max pool ----
    zero_f32_kernel<<<outz_b, 256, 0, stream>>>(out, (long)N_GRAPHS_C * 256);
    pool_max_kernel<<<pool_b, 256, 0, stream>>>(x1, x2, batch, out, NN);
}
